// TemporalSelfAttentionLayer_70858370449757
// MI455X (gfx1250) — matrix operand order repaired and checked by reading, not compiled, not run on hardware
//
#include <hip/hip_runtime.h>
#include <hip/hip_bf16.h>

typedef __attribute__((ext_vector_type(16))) _Float16 v16h;
typedef __attribute__((ext_vector_type(8)))  _Float16 v8h;
typedef __attribute__((ext_vector_type(8)))  float    v8f;

namespace mha455 {

constexpr int SEQ = 4096;
constexpr int CH  = 256;
constexpr int NH  = 4;
constexpr int HD  = 64;
constexpr int NB  = 2;
constexpr int M   = NB * SEQ;                      // 8192 rows (B*N)
constexpr float QSCALE = 0.125f * 1.4426950408889634f;  // D^-0.5 * log2(e)

__device__ __forceinline__ v8f vzero8f() {
  v8f z;
#pragma unroll
  for (int i = 0; i < 8; ++i) z[i] = 0.0f;
  return z;
}

__device__ __forceinline__ v16h cat16(v8h lo, v8h hi) {
  v16h r;
#pragma unroll
  for (int i = 0; i < 8; ++i) { r[i] = lo[i]; r[i + 8] = hi[i]; }
  return r;
}

// A fragment (16x32 f16, row-major source, pitch in halfs).
// lane l: l15=l&15 is row M; element e -> K = (e<8 ? e+8h : e+8+8h),
// i.e. two contiguous 16B reads per lane at K offsets 8h and 16+8h.
__device__ __forceinline__ v16h load_a16x32(const _Float16* p, int pitch, int l15, int hf) {
  const _Float16* row = p + (size_t)l15 * pitch + 8 * hf;
  v8h lo = *(const v8h*)(row);
  v8h hi = *(const v8h*)(row + 16);
  return cat16(lo, hi);
}

// B fragment (32x16 f16) from N-major storage bT[n][k] (pitch in halfs).
// lane l: col N = l15; element e -> K = (e<8 ? e+8h : e+8+8h) (matches A).
__device__ __forceinline__ v16h load_b32x16_t(const _Float16* bT, size_t pitch, int l15, int hf) {
  const _Float16* row = bT + (size_t)l15 * pitch + 8 * hf;
  v8h lo = *(const v8h*)(row);
  v8h hi = *(const v8h*)(row + 16);
  return cat16(lo, hi);
}

__device__ __forceinline__ v8f wmma_f16(v16h a, v16h b, v8f c) {
  // D = A*B + C, f32 accumulate
  return __builtin_amdgcn_wmma_f32_16x16x32_f16(false, a, false, b, (short)0, c, false, false);
}

} // namespace mha455

using namespace mha455;

// ---------------------------------------------------------------- converts
__global__ void mha_cvt_f16(const float* __restrict__ src,
                            _Float16* __restrict__ dst, int n) {
  int i = blockIdx.x * blockDim.x + threadIdx.x;
  const int stride = gridDim.x * blockDim.x;
  for (; i < n; i += stride) dst[i] = (_Float16)src[i];
}

// ---------------------------------------------------------------- QKV GEMM
// qkv = x @ qkv_w^T + b. Q (pre-scaled by QSCALE) and K go to [BH][SEQ][HD];
// V goes to TRANSPOSED [BH][HD][SEQ] so attention's PV B-fragments are
// contiguous global reads. The transposed V store is a single v8h per frag:
// each lane holds 8 consecutive rows (n) of one column (d).
__global__ __launch_bounds__(128) void mha_qkv_gemm(
    const _Float16* __restrict__ xh,    // [M][CH]
    const _Float16* __restrict__ wh,    // [3CH][CH]  (N-major for B frags)
    const float*    __restrict__ bias,  // [3CH]
    _Float16* __restrict__ qh,          // [BH][SEQ][HD]
    _Float16* __restrict__ kh,          // [BH][SEQ][HD]
    _Float16* __restrict__ vth) {       // [BH][HD][SEQ]
  const int wave = threadIdx.x >> 5;
  const int lane = threadIdx.x & 31;
  const int l15 = lane & 15, hf = lane >> 4;
  const int row0 = blockIdx.x * 64 + wave * 16;
  const int col0 = blockIdx.y * 64;

  v8f acc[4] = {vzero8f(), vzero8f(), vzero8f(), vzero8f()};

  for (int k0 = 0; k0 < CH; k0 += 32) {
    v16h a = load_a16x32(xh + (size_t)row0 * CH + k0, CH, l15, hf);
#pragma unroll
    for (int c = 0; c < 4; ++c) {
      v16h b = load_b32x16_t(wh + (size_t)(col0 + c * 16) * CH + k0, CH, l15, hf);
      acc[c] = wmma_f16(a, b, acc[c]);
    }
  }

  const int bidx = row0 >> 12;              // batch (uniform: row tile never
  const int n0   = (row0 & (SEQ - 1));      //  straddles a 4096 boundary)
#pragma unroll
  for (int c = 0; c < 4; ++c) {
    const int j = col0 + c * 16 + l15;      // output column in [0,768)
    const float bj = bias[j];
    const int type = j >> 8;                // 0=Q 1=K 2=V (uniform per frag)
    const int head = (j >> 6) & 3;
    const int d    = j & 63;
    if (type == 2) {
      v8h pack;
#pragma unroll
      for (int r = 0; r < 8; ++r) pack[r] = (_Float16)(acc[c][r] + bj);
      *(v8h*)(vth + ((size_t)(bidx * NH + head) * HD + d) * SEQ + n0 + 8 * hf) = pack;
    } else {
      _Float16* dst = (type == 0) ? qh : kh;
      const float sc = (type == 0) ? QSCALE : 1.0f;
#pragma unroll
      for (int r = 0; r < 8; ++r) {
        const int n = n0 + r + 8 * hf;
        dst[((size_t)(bidx * NH + head) * SEQ + n) * HD + d] =
            (_Float16)((acc[c][r] + bj) * sc);
      }
    }
  }
}

// ---------------------------------------------------------------- flash attention
// grid: (SEQ/128, NB*NH), block 256 = 8 independent waves; each wave owns
// 16 q rows x full D. No block barriers: K and V^T B-fragments stream from
// global (L2/WGP$-resident, 0.5 MB per (b,h)); only the P D->A layout
// roundtrip uses per-wave LDS. 64-key tiles, 16 WMMAs per tile per wave.
__global__ __launch_bounds__(256) void mha_flash_attn(
    const _Float16* __restrict__ qh,    // [BH][SEQ][HD], pre-scaled
    const _Float16* __restrict__ kh,    // [BH][SEQ][HD]
    const _Float16* __restrict__ vth,   // [BH][HD][SEQ]
    _Float16* __restrict__ aoh) {       // [M][CH] f16 (b, n, h*64+d)
  constexpr int PP = 72;                // P staging pitch (144B rows)
  __shared__ __align__(16) _Float16 Ps[8][16 * PP];

  const int wave = threadIdx.x >> 5;
  const int lane = threadIdx.x & 31;
  const int l15 = lane & 15, hf = lane >> 4;
  const int bh = blockIdx.y;
  const int q0 = blockIdx.x * 128 + wave * 16;

  const _Float16* Qp  = qh  + (size_t)bh * SEQ * HD + (size_t)q0 * HD;
  const _Float16* Kp  = kh  + (size_t)bh * SEQ * HD;
  const _Float16* Vtp = vth + (size_t)bh * HD * SEQ;

  const v16h aq0 = load_a16x32(Qp, HD, l15, hf);        // d 0..31
  const v16h aq1 = load_a16x32(Qp + 32, HD, l15, hf);   // d 32..63

  v8f o[4] = {vzero8f(), vzero8f(), vzero8f(), vzero8f()};
  float mrow[8], lrow[8];
#pragma unroll
  for (int i = 0; i < 8; ++i) { mrow[i] = -__builtin_inff(); lrow[i] = 0.0f; }

  _Float16* Pw = &Ps[wave][0];

  for (int kb = 0; kb < SEQ; kb += 64) {
    // prefetch next tile into the cache hierarchy (global_prefetch_b8)
    if (kb + 64 < SEQ) {
      __builtin_prefetch(Kp  + (size_t)(kb + 64 + 2 * lane) * HD, 0, 3);
      __builtin_prefetch(Vtp + (size_t)(2 * lane) * SEQ + kb + 64, 0, 3);
    }

    // S = Q K^T for 64 keys (four 16-col sub-tiles), f32 accumulate
    v8f s[4];
#pragma unroll
    for (int c = 0; c < 4; ++c) {
      const _Float16* kt = Kp + (size_t)(kb + c * 16) * HD;
      s[c] = wmma_f16(aq0, load_b32x16_t(kt, HD, l15, hf), vzero8f());
      s[c] = wmma_f16(aq1, load_b32x16_t(kt + 32, HD, l15, hf), s[c]);
    }

    // online softmax, base-2 (scale*log2e folded into Q). Row r+8*hf lives
    // in one 16-lane half -> xor masks 1,2,4,8 reduce the row.
#pragma unroll
    for (int r = 0; r < 8; ++r) {
      const float v0 = s[0][r], v1 = s[1][r], v2 = s[2][r], v3 = s[3][r];
      float mx = fmaxf(fmaxf(v0, v1), fmaxf(v2, v3));
      mx = fmaxf(mx, __shfl_xor(mx, 1));
      mx = fmaxf(mx, __shfl_xor(mx, 2));
      mx = fmaxf(mx, __shfl_xor(mx, 4));
      mx = fmaxf(mx, __shfl_xor(mx, 8));
      const float mn = fmaxf(mrow[r], mx);
      const float corr = __builtin_exp2f(mrow[r] - mn);
      const float p0 = __builtin_exp2f(v0 - mn);
      const float p1 = __builtin_exp2f(v1 - mn);
      const float p2 = __builtin_exp2f(v2 - mn);
      const float p3 = __builtin_exp2f(v3 - mn);
      float rs = (p0 + p1) + (p2 + p3);
      rs += __shfl_xor(rs, 1);
      rs += __shfl_xor(rs, 2);
      rs += __shfl_xor(rs, 4);
      rs += __shfl_xor(rs, 8);
      lrow[r] = lrow[r] * corr + rs;
      mrow[r] = mn;
      o[0][r] *= corr; o[1][r] *= corr; o[2][r] *= corr; o[3][r] *= corr;
      // D-fragment layout -> LDS tile [16][64] so P can re-enter as A frags
      _Float16* pr = Pw + (r + 8 * hf) * PP + l15;
      pr[0]  = (_Float16)p0;
      pr[16] = (_Float16)p1;
      pr[32] = (_Float16)p2;
      pr[48] = (_Float16)p3;
    }

    const v16h ap0 = load_a16x32(Pw, PP, l15, hf);        // keys kb+0..31
    const v16h ap1 = load_a16x32(Pw + 32, PP, l15, hf);   // keys kb+32..63

    // O += P V  : V^T rows are d, cols are keys -> contiguous B frags
#pragma unroll
    for (int d4 = 0; d4 < 4; ++d4) {
      const _Float16* vt = Vtp + (size_t)(d4 * 16) * SEQ + kb;
      o[d4] = wmma_f16(ap0, load_b32x16_t(vt, SEQ, l15, hf), o[d4]);
      o[d4] = wmma_f16(ap1, load_b32x16_t(vt + 32, SEQ, l15, hf), o[d4]);
    }
  }

  // normalize and scatter to [b][n][h*64+d]
  const int b = bh >> 2, head = bh & 3;
#pragma unroll
  for (int d4 = 0; d4 < 4; ++d4) {
    const int col = head * HD + d4 * 16 + l15;
#pragma unroll
    for (int r = 0; r < 8; ++r) {
      const int n = q0 + r + 8 * hf;
      aoh[((size_t)(b * SEQ + n)) * CH + col] = (_Float16)(o[d4][r] / lrow[r]);
    }
  }
}

// ---------------------------------------------------------------- output projection
__global__ __launch_bounds__(128) void mha_out_proj(
    const _Float16* __restrict__ ah,    // [M][CH]
    const _Float16* __restrict__ wh,    // [CH][CH] (N-major)
    const float*    __restrict__ bias,  // [CH]
    float* __restrict__ out) {          // [M][CH] f32
  const int wave = threadIdx.x >> 5;
  const int lane = threadIdx.x & 31;
  const int l15 = lane & 15, hf = lane >> 4;
  const int row0 = blockIdx.x * 64 + wave * 16;
  const int col0 = blockIdx.y * 64;

  v8f acc[4] = {vzero8f(), vzero8f(), vzero8f(), vzero8f()};
  for (int k0 = 0; k0 < CH; k0 += 32) {
    v16h a = load_a16x32(ah + (size_t)row0 * CH + k0, CH, l15, hf);
#pragma unroll
    for (int c = 0; c < 4; ++c) {
      v16h b = load_b32x16_t(wh + (size_t)(col0 + c * 16) * CH + k0, CH, l15, hf);
      acc[c] = wmma_f16(a, b, acc[c]);
    }
  }
#pragma unroll
  for (int c = 0; c < 4; ++c) {
    const int col = col0 + c * 16 + l15;
    const float bj = bias[col];
#pragma unroll
    for (int r = 0; r < 8; ++r) {
      out[(size_t)(row0 + r + 8 * hf) * CH + col] = acc[c][r] + bj;
    }
  }
}

// ---------------------------------------------------------------- launch
extern "C" void kernel_launch(void* const* d_in, const int* in_sizes, int n_in,
                              void* d_out, int out_size, void* d_ws, size_t ws_size,
                              hipStream_t stream) {
  (void)in_sizes; (void)n_in; (void)out_size; (void)ws_size;
  const float* x     = (const float*)d_in[0];
  const float* qkv_w = (const float*)d_in[1];
  const float* qkv_b = (const float*)d_in[2];
  const float* out_w = (const float*)d_in[3];
  const float* out_b = (const float*)d_in[4];

  char* ws = (char*)d_ws;
  _Float16* xh  = (_Float16*)ws; ws += (size_t)M * CH * 2;          // 4 MB
  _Float16* qwh = (_Float16*)ws; ws += (size_t)3 * CH * CH * 2;     // 384 KB
  _Float16* owh = (_Float16*)ws; ws += (size_t)CH * CH * 2;         // 128 KB
  _Float16* qh  = (_Float16*)ws; ws += (size_t)NB * NH * SEQ * HD * 2;
  _Float16* kh  = (_Float16*)ws; ws += (size_t)NB * NH * SEQ * HD * 2;
  _Float16* vth = (_Float16*)ws; ws += (size_t)NB * NH * HD * SEQ * 2;
  _Float16* aoh = (_Float16*)ws; ws += (size_t)M * CH * 2;

  mha_cvt_f16<<<512, 256, 0, stream>>>(x, xh, M * CH);
  mha_cvt_f16<<<192, 256, 0, stream>>>(qkv_w, qwh, 3 * CH * CH);
  mha_cvt_f16<<<64, 256, 0, stream>>>(out_w, owh, CH * CH);

  mha_qkv_gemm<<<dim3(M / 64, (3 * CH) / 64), 128, 0, stream>>>(
      xh, qwh, qkv_b, qh, kh, vth);

  mha_flash_attn<<<dim3(SEQ / 128, NB * NH), 256, 0, stream>>>(qh, kh, vth, aoh);

  mha_out_proj<<<dim3(M / 64, CH / 64), 128, 0, stream>>>(
      aoh, owh, out_b, (float*)d_out);
}
